// OriOrthLinear_16192026705942
// MI455X (gfx1250) — compile-verified, matrix operand order checked by reading, not run on hardware
//
#include <hip/hip_runtime.h>
#include <hip/hip_bf16.h>

#define GROUPS 8
#define BS     4096
#define CH     512
#define BK     128            // reflections per WY panel
#define NB     (CH / BK)      // 4 panels
#define TILE_M 128            // rows of x per workgroup
#define KC     64             // K-extent of one staged B slab (2 WMMA k-steps)
#define BLD2   72             // padded LDS row stride for B slabs (bank spread)
#define PSLD   136            // padded LDS row stride for P (bank spread)

typedef __attribute__((ext_vector_type(16))) __bf16 v16bf;
typedef __attribute__((ext_vector_type(8)))  __bf16 v8bf;
typedef __attribute__((ext_vector_type(8)))  float  v8f;
typedef __attribute__((ext_vector_type(4)))  float  v4f;
typedef __attribute__((ext_vector_type(4)))  int    v4i;

#if defined(__AMDGCN__) && __has_builtin(__builtin_amdgcn_global_load_async_to_lds_b128)
#define ASYNC_LDS 1
#else
#define ASYNC_LDS 0
#endif

#if defined(__AMDGCN__) && __has_builtin(__builtin_amdgcn_sched_group_barrier)
#define SGB(mask, size) __builtin_amdgcn_sched_group_barrier(mask, size, 0)
#else
#define SGB(mask, size)
#endif

// 16-byte global -> LDS copy (async on gfx1250 when available)
__device__ __forceinline__ void cp16(const __bf16* g, __bf16* l) {
#if ASYNC_LDS
  __builtin_amdgcn_global_load_async_to_lds_b128(
      (__attribute__((address_space(1))) v4i*)g,
      (__attribute__((address_space(3))) v4i*)l, 0, 0);
#else
  *(v8bf*)l = *(const v8bf*)g;
#endif
}

__device__ __forceinline__ void cp_wait() {
#if ASYNC_LDS
#if __has_builtin(__builtin_amdgcn_s_wait_asynccnt)
  __builtin_amdgcn_s_wait_asynccnt(0);
#else
  asm volatile("s_wait_asynccnt 0" ::: "memory");
#endif
#endif
}

// Stage one 128(n) x 64(k) bf16 slab into LDS (padded stride BLD2); 256 threads.
__device__ __forceinline__ void stage_slab(const __bf16* __restrict__ g, size_t gstride,
                                           __bf16* __restrict__ l, int tid) {
#pragma unroll
  for (int p = 0; p < 4; ++p) {
    int e  = tid + p * 256;        // 0..1023 : 128 rows x 8 sixteen-byte pieces
    int n  = e >> 3;
    int c8 = (e & 7) * 8;
    cp16(g + (size_t)n * gstride + c8, l + n * BLD2 + c8);
  }
}

// Gather one 16x32 bf16 WMMA operand fragment: two contiguous 8-elem chunks.
__device__ __forceinline__ v16bf frag16(const __bf16* base) {
  v8bf b0 = *(const v8bf*)(base);
  v8bf b1 = *(const v8bf*)(base + 16);
  v16bf b;
#pragma unroll
  for (int i = 0; i < 8; ++i) { b[i] = b0[i]; b[8 + i] = b1[i]; }
  return b;
}

// Issue raw f32 loads for one 64-wide A stage (2 fragments worth).
__device__ __forceinline__ void load_raw(const float* __restrict__ xb, int k0, v4f* r) {
#pragma unroll
  for (int h = 0; h < 2; ++h) {
    const float* p = xb + k0 + h * 32;
    r[h * 4 + 0] = *(const v4f*)(p);
    r[h * 4 + 1] = *(const v4f*)(p + 4);
    r[h * 4 + 2] = *(const v4f*)(p + 16);
    r[h * 4 + 3] = *(const v4f*)(p + 20);
  }
}

// Convert 4 raw v4f (one fragment) to a bf16 A fragment.
__device__ __forceinline__ v16bf cvt_a(const v4f* r) {
  v16bf a;
#pragma unroll
  for (int i = 0; i < 4; ++i) {
    a[i]      = (__bf16)r[0][i];  a[4 + i]  = (__bf16)r[1][i];
    a[8 + i]  = (__bf16)r[2][i];  a[12 + i] = (__bf16)r[3][i];
  }
  return a;
}

// ---------------------------------------------------------------------------
// Kernel 1: per (group, panel) Gram matrix  G[i][j] = v_{j0+i} . v_{j0+j}
// ---------------------------------------------------------------------------
__global__ __launch_bounds__(256)
void gram_kernel(const float* __restrict__ W, float* __restrict__ G) {
  int wg = blockIdx.x;              // g*NB + jb
  int g  = wg / NB, jb = wg % NB;
  int j0 = jb * BK;
  const float* Wg = W + (size_t)g * CH * CH;
  float* Gb = G + (size_t)wg * BK * BK;
  for (int e = threadIdx.x; e < BK * BK; e += blockDim.x) {
    int i = e / BK, j = e % BK;
    float s = 0.f;
    for (int c = 0; c < CH; ++c)
      s += Wg[(size_t)c * CH + j0 + i] * Wg[(size_t)c * CH + j0 + j];
    Gb[e] = s;
  }
}

// ---------------------------------------------------------------------------
// Kernel 2: build T_b (compact WY, transposed in LDS) then emit
//   Mt[c][k]  = (T_b V_b^T)[k][c]   (bf16, column-major for WMMA B frags)
//   Vcol[n][c] = V_b[c][n]          (bf16, column-major for WMMA B frags)
// ---------------------------------------------------------------------------
__global__ __launch_bounds__(256)
void factor_kernel(const float* __restrict__ W, const float* __restrict__ G,
                   __bf16* __restrict__ Mt, __bf16* __restrict__ Vcol) {
  __shared__ float Tt[BK * BK];     // T transposed: Tt[k*BK + r] = T[r][k]; 64 KB
  int wg = blockIdx.x;
  int g  = wg / NB, jb = wg % NB;
  int j0 = jb * BK;
  const float* Wg = W + (size_t)g * CH * CH;
  const float* Gb = G + (size_t)wg * BK * BK;

  for (int e = threadIdx.x; e < BK * BK; e += blockDim.x) Tt[e] = 0.f;
  __syncthreads();

  // Sequential column recurrence: T[:j,j] = -beta_j * T[:j,:j] @ G[:j,j]
  for (int j = 0; j < BK; ++j) {
    float beta = 2.0f / Gb[j * BK + j];
    int r = threadIdx.x;
    if (r < j) {
      float s = 0.f;
      for (int k = r; k < j; ++k) s += Tt[k * BK + r] * Gb[k * BK + j];
      Tt[j * BK + r] = -beta * s;
    } else if (r == j) {
      Tt[j * BK + j] = beta;
    }
    __syncthreads();
  }

  // Mt[c][k] = sum_kk T[k][kk] * W[c][j0+kk]
  __bf16* Mb = Mt + (size_t)wg * CH * BK;
  for (int e = threadIdx.x; e < CH * BK; e += blockDim.x) {
    int c = e / BK, k = e % BK;
    float s = 0.f;
    for (int kk = 0; kk < BK; ++kk)
      s += Tt[kk * BK + k] * Wg[(size_t)c * CH + j0 + kk];
    Mb[e] = (__bf16)s;
  }
  // Vcol[n][c] = W[c][j0+n]
  __bf16* Vb = Vcol + (size_t)wg * BK * CH;
  for (int e = threadIdx.x; e < BK * CH; e += blockDim.x) {
    int n = e / CH, c = e % CH;
    Vb[e] = (__bf16)Wg[(size_t)c * CH + j0 + n];
  }
}

// ---------------------------------------------------------------------------
// Kernel 3 (one launch per panel, sequential): x <- x - (x V_b)(T_b V_b^T)
// Workgroup = 128 rows x 512 cols of one group; 8 waves, wave = 16-row strip.
// B operands staged through double-buffered 64-wide LDS slabs (async copies
// shared by all 8 waves, one barrier per 16 WMMAs); P staged bf16 in LDS.
// sched_group_barrier hints pipeline DS-reads 4 fragments ahead of the WMMAs.
// ---------------------------------------------------------------------------
__global__ __launch_bounds__(256)
void apply_kernel(const float* __restrict__ src, float* __restrict__ dst,
                  const __bf16* __restrict__ Vcol, const __bf16* __restrict__ Mt,
                  int jb) {
  __shared__ __align__(16) __bf16 Bbuf[2][BK * BLD2];  // 2 x 18 KB B slabs
  __shared__ __align__(16) __bf16 Ps[TILE_M * PSLD];   // 34 KB staged P
  int g    = blockIdx.y;
  int m0   = blockIdx.x * TILE_M;
  int tid  = threadIdx.x;
  int wave = tid >> 5, lane = tid & 31;
  int lrow = lane & 15, khalf = lane >> 4;
  int wg   = g * NB + jb;
  const __bf16* Vb = Vcol + (size_t)wg * BK * CH;
  const __bf16* Mb = Mt   + (size_t)wg * CH * BK;
  const v8f vzero = {0.f, 0.f, 0.f, 0.f, 0.f, 0.f, 0.f, 0.f};

  // ---- GEMM1: P[strip] = X[strip] * V_b  (16x128 per wave, K = 512) ----
  int rowA = m0 + wave * 16 + lrow;
  const float* xb = src + ((size_t)g * BS + rowA) * CH + khalf * 8;

  stage_slab(Vb, CH, Bbuf[0], tid);
  v16bf a0, a1;
  {
    v4f r[8];
    load_raw(xb, 0, r);
    a0 = cvt_a(r); a1 = cvt_a(r + 4);
  }
  cp_wait();
  __syncthreads();

  v8f acc[8] = {};
  for (int s = 0; s < CH / KC; ++s) {          // 8 stages of K=64
    v4f r[8];
    if (s + 1 < CH / KC) {
      stage_slab(Vb + (s + 1) * KC, CH, Bbuf[(s + 1) & 1], tid);
      load_raw(xb, (s + 1) * KC, r);           // raw loads in flight over WMMAs
    }
#pragma unroll
    for (int half = 0; half < 2; ++half) {
      const __bf16* bb = Bbuf[s & 1] + (size_t)lrow * BLD2 + half * 32 + khalf * 8;
      v16bf af = half ? a1 : a0;
#pragma unroll
      for (int np = 0; np < 2; ++np) {         // batches of 4 B frags
        v16bf bf[4];
#pragma unroll
        for (int j = 0; j < 4; ++j)
          bf[j] = frag16(bb + (size_t)((np * 4 + j) * 16) * BLD2);
#pragma unroll
        for (int j = 0; j < 4; ++j)
          acc[np * 4 + j] = __builtin_amdgcn_wmma_f32_16x16x32_bf16(
              false, af, false, bf[j], (short)0, acc[np * 4 + j], false, false);
      }
    }
    // pipeline hint: keep 4 fragments (8 DS reads) ahead of each 4-WMMA batch
#pragma unroll
    for (int q = 0; q < 4; ++q) { SGB(0x100, 8); SGB(0x008, 4); }
    cp_wait();
    __syncthreads();
    if (s + 1 < CH / KC) {                     // convert after latency is hidden
      a0 = cvt_a(r); a1 = cvt_a(r + 4);
    }
  }

  // stash P as bf16 in LDS (C-layout: lane holds N=lrow, rows v + 8*khalf)
#pragma unroll
  for (int nt = 0; nt < 8; ++nt)
#pragma unroll
    for (int v = 0; v < 8; ++v)
      Ps[(wave * 16 + v + 8 * khalf) * PSLD + nt * 16 + lrow] = (__bf16)acc[nt][v];

  // prefetch first M_b slab (nc=0, k-slab 0) while P stores drain
  stage_slab(Mb, BK, Bbuf[0], tid);
  cp_wait();
  __syncthreads();

  // ---- GEMM2: D[strip] = P[strip] * M_b ; dst = src - D  (K = 128) ----
  const __bf16* prow = Ps + (size_t)(wave * 16 + lrow) * PSLD;
  v8f acc2[8] = {};
  for (int t = 0; t < 8; ++t) {                // nc = t>>1, k-slab sh = t&1
    int nc = t >> 1, sh = t & 1;
    if (t + 1 < 8) {
      int nc2 = (t + 1) >> 1, sh2 = (t + 1) & 1;
      stage_slab(Mb + ((size_t)nc2 * BK) * BK + sh2 * KC, BK, Bbuf[(t + 1) & 1], tid);
    }
    v16bf p0 = frag16(prow + (sh * 2 + 0) * 32 + khalf * 8);
    v16bf p1 = frag16(prow + (sh * 2 + 1) * 32 + khalf * 8);
#pragma unroll
    for (int half = 0; half < 2; ++half) {
      const __bf16* bb = Bbuf[t & 1] + (size_t)lrow * BLD2 + half * 32 + khalf * 8;
      v16bf af = half ? p1 : p0;
#pragma unroll
      for (int np = 0; np < 2; ++np) {
        v16bf bf[4];
#pragma unroll
        for (int j = 0; j < 4; ++j)
          bf[j] = frag16(bb + (size_t)((np * 4 + j) * 16) * BLD2);
#pragma unroll
        for (int j = 0; j < 4; ++j)
          acc2[np * 4 + j] = __builtin_amdgcn_wmma_f32_16x16x32_bf16(
              false, af, false, bf[j], (short)0, acc2[np * 4 + j], false, false);
      }
    }
    // pipeline hint: P frags (4 DS reads) first, then (8 DS reads, 4 WMMAs) x4
    SGB(0x100, 4);
#pragma unroll
    for (int q = 0; q < 4; ++q) { SGB(0x100, 8); SGB(0x008, 4); }
    if (sh == 1) {   // finished this 128-wide column chunk: fuse subtraction
#pragma unroll
      for (int nt = 0; nt < 8; ++nt) {
#pragma unroll
        for (int v = 0; v < 8; ++v) {
          int row = m0 + wave * 16 + v + 8 * khalf;
          int col = nc * BK + nt * 16 + lrow;
          size_t idx = ((size_t)g * BS + row) * CH + col;
          dst[idx] = src[idx] - acc2[nt][v];
        }
        acc2[nt] = vzero;
      }
    }
    cp_wait();
    __syncthreads();
  }
}

// ---------------------------------------------------------------------------
extern "C" void kernel_launch(void* const* d_in, const int* in_sizes, int n_in,
                              void* d_out, int out_size, void* d_ws, size_t ws_size,
                              hipStream_t stream) {
  const float* x = (const float*)d_in[0];   // (8, 4096, 512) f32
  const float* w = (const float*)d_in[1];   // (8, 512, 512)  f32
  float* out = (float*)d_out;

  char* ws = (char*)d_ws;
  size_t gBytes = (size_t)GROUPS * NB * BK * BK * sizeof(float);   // 2 MB
  size_t mBytes = (size_t)GROUPS * NB * CH * BK * sizeof(__bf16);  // 4 MB
  float*  G    = (float*)ws;
  __bf16* Mt   = (__bf16*)(ws + gBytes);
  __bf16* Vcol = (__bf16*)(ws + gBytes + mBytes);                  // +4 MB

  gram_kernel  <<<GROUPS * NB, 256, 0, stream>>>(w, G);
  factor_kernel<<<GROUPS * NB, 256, 0, stream>>>(w, G, Mt, Vcol);
  for (int jb = 0; jb < NB; ++jb) {
    const float* s = (jb == 0) ? x : out;
    apply_kernel<<<dim3(BS / TILE_M, GROUPS), 256, 0, stream>>>(s, out, Vcol, Mt, jb);
  }
}
